// cross_attention_89653147337302
// MI455X (gfx1250) — matrix operand order repaired and checked by reading, not compiled, not run on hardware
//
#include <hip/hip_runtime.h>
#include <hip/hip_bf16.h>

// Problem dims (fixed by reference): b=16, c=64, n=h*w=4096
static constexpr int BB = 16;
static constexpr int CC = 64;
static constexpr int NN = 4096;
static constexpr float SCALE = 50.0f;
// exp(50*(s-1)) == exp2(s*K - K), K = 50*log2(e): single FMA + v_exp_f32
static constexpr float K2 = 50.0f * 1.4426950408889634f;

typedef __attribute__((ext_vector_type(16))) __bf16 v16bf;
typedef __attribute__((ext_vector_type(8)))  float  v8f;

union QV { uint4 q[2]; v16bf v; };
union BF2 { __bf16 h[2]; unsigned int u; };

// ---------------------------------------------------------------------------
// Kernel 1: per-(b,n) L2 normalization over c, pack to bf16 row-major [b][n][c].
// Strided fp32 reads are coalesced along n; bf16 rows are staged in padded LDS
// so the write-out is contiguous coalesced dwords (block's 256 rows = 32 KB
// contiguous span of the [n][c] layout).
// ---------------------------------------------------------------------------
__global__ void nk_normalize_pack(const float* __restrict__ xv,
                                  const float* __restrict__ xt,
                                  __bf16* __restrict__ nv,
                                  __bf16* __restrict__ nt)
{
    __shared__ unsigned int sbuf[256][33];                 // 32 payload + 1 pad (bank spread)
    const int t  = threadIdx.x;                            // 0..255 -> one n each
    const int n0 = blockIdx.x * 256;
    const int b  = blockIdx.y;
    const float* src = blockIdx.z ? xt : xv;
    __bf16*      dst = blockIdx.z ? nt : nv;

    const float* col = src + (size_t)b * CC * NN + (n0 + t);
    float ss = 0.f;
#pragma unroll
    for (int c = 0; c < CC; ++c) { float v = col[(size_t)c * NN]; ss += v * v; }
    const float inv = 1.0f / fmaxf(sqrtf(ss), 1e-12f);     // F.normalize eps

#pragma unroll
    for (int c = 0; c < CC; c += 2) {
        BF2 p;
        p.h[0] = (__bf16)(col[(size_t)c * NN] * inv);
        p.h[1] = (__bf16)(col[(size_t)(c + 1) * NN] * inv);
        sbuf[t][c >> 1] = p.u;
    }
    __syncthreads();

    unsigned int* out = (unsigned int*)(dst + ((size_t)b * NN + n0) * CC);
#pragma unroll
    for (int k = t; k < 256 * 32; k += 256)                // coalesced dword stores
        out[k] = sbuf[k >> 5][k & 31];
}

// ---------------------------------------------------------------------------
// Kernel 2: flash row-stats.  4 waves/block; each wave owns 64 rows of sim
// (four 16-row A tiles in registers) and streams 256 B tiles: 8 WMMAs per
// B-tile load (4x B reuse).  launch_bounds(128,1) + waves_per_eu(1): this
// kernel trades occupancy for a fat register file -- the working set
// (A 64 + acc 32 + rsum 32 + B 16 VGPRs) must stay in registers; the 256
// independent m-iterations provide all the latency hiding needed.
//
// Softmax shift-invariance: sim = <unit,unit> so |sim| <= ~1.01 after bf16
// rounding; the fixed shift 1.0 replaces the row max exactly, so the inner
// loop has NO cross-lane ops: rsum += exp2(s*K - K).  The 16-lane row
// reduction happens once, after the m-loop.
// ---------------------------------------------------------------------------
__device__ __forceinline__ float rowsum16(float v) {
    v += __shfl_xor(v, 1, 32);
    v += __shfl_xor(v, 2, 32);
    v += __shfl_xor(v, 4, 32);
    v += __shfl_xor(v, 8, 32);
    return v;
}

__global__ void
__launch_bounds__(128, 1)
__attribute__((amdgpu_waves_per_eu(1)))
nk_flash_rowstats(const __bf16* __restrict__ nv,
                  const __bf16* __restrict__ nt,
                  float* __restrict__ rel_mean)
{
    const int lane = threadIdx.x & 31;
    const int wid  = threadIdx.x >> 5;                     // 0..3
    const int lm   = lane & 15;
    const int half = lane >> 4;
    const int b    = blockIdx.y;
    const int n0   = blockIdx.x * 256 + wid * 64;          // this wave's 64 rows

    // A tiles: 16x32 bf16 layout -> lane lm(+16) holds M=lm,
    // K = {h*8..h*8+7, 16+h*8..16+h*8+7}; contiguous 16B chunks per lane.
    QV a[4][2];
#pragma unroll
    for (int j = 0; j < 4; ++j) {
        const char* Arow =
            (const char*)(nv + ((size_t)b * NN + n0 + j * 16 + lm) * CC);
        a[j][0].q[0] = *(const uint4*)(Arow +      half * 16);
        a[j][0].q[1] = *(const uint4*)(Arow + 32 + half * 16);
        a[j][1].q[0] = *(const uint4*)(Arow + 64 + half * 16);
        a[j][1].q[1] = *(const uint4*)(Arow + 96 + half * 16);
    }

    float rsum[4][8];
#pragma unroll
    for (int j = 0; j < 4; ++j)
#pragma unroll
        for (int i = 0; i < 8; ++i) rsum[j][i] = 0.f;

    const __bf16* Bbase = nt + (size_t)b * NN * CC;
    for (int m0 = 0; m0 < NN; m0 += 16) {
        // B 32x16 bf16 layout: lane lm(+16) holds N=lm,
        // K = {h*8..h*8+7, 16+h*8..16+h*8+7} (same order as A).
        const char* Brow = (const char*)(Bbase + (size_t)(m0 + lm) * CC);
        QV b0, b1;
        b0.q[0] = *(const uint4*)(Brow +      half * 16);
        b0.q[1] = *(const uint4*)(Brow + 32 + half * 16);
        b1.q[0] = *(const uint4*)(Brow + 64 + half * 16);
        b1.q[1] = *(const uint4*)(Brow + 96 + half * 16);

        v8f acc[4];
#pragma unroll
        for (int j = 0; j < 4; ++j) {
            v8f z = {};
            z = __builtin_amdgcn_wmma_f32_16x16x32_bf16(
                    false, a[j][0].v, false, b0.v, (short)0, z, false, false);
            z = __builtin_amdgcn_wmma_f32_16x16x32_bf16(
                    false, a[j][1].v, false, b1.v, (short)0, z, false, false);
            acc[j] = z;
        }
#pragma unroll
        for (int j = 0; j < 4; ++j)
#pragma unroll
            for (int i = 0; i < 8; ++i)
                rsum[j][i] += __builtin_amdgcn_exp2f(fmaf(acc[j][i], K2, -K2));
    }

    // mean(relation_row) = (sum exp / Z) / NN with Z == sum exp -> (l/l)/NN
    // (exact value of the reference expression; l > 0 always).
#pragma unroll
    for (int j = 0; j < 4; ++j)
#pragma unroll
        for (int i = 0; i < 8; ++i) {
            const float rs = rowsum16(rsum[j][i]);
            if (lm == i) {
                const int row = n0 + j * 16 + i + half * 8;
                rel_mean[(size_t)b * NN + row] = (rs / rs) * (1.0f / (float)NN);
            }
        }
}

// ---------------------------------------------------------------------------
// Kernel 3: weight = softmax(rel_mean, per-batch) + 1        (16 x 4096, tiny)
// ---------------------------------------------------------------------------
__global__ void nk_weight(const float* __restrict__ rel_mean,
                          float* __restrict__ weight)
{
    __shared__ float red[256];
    const int b = blockIdx.x, t = threadIdx.x;
    const float* r = rel_mean + (size_t)b * NN;

    float lmax = -__builtin_inff();
    for (int i = t; i < NN; i += 256) lmax = fmaxf(lmax, r[i]);
    red[t] = lmax; __syncthreads();
    for (int s = 128; s > 0; s >>= 1) {
        if (t < s) red[t] = fmaxf(red[t], red[t + s]);
        __syncthreads();
    }
    const float mx = red[0]; __syncthreads();

    float lsum = 0.f;
    for (int i = t; i < NN; i += 256) lsum += __expf(r[i] - mx);
    red[t] = lsum; __syncthreads();
    for (int s = 128; s > 0; s >>= 1) {
        if (t < s) red[t] += red[t + s];
        __syncthreads();
    }
    const float inv = 1.0f / red[0];

    for (int i = t; i < NN; i += 256)
        weight[(size_t)b * NN + i] = __expf(r[i] - mx) * inv + 1.0f;
}

// ---------------------------------------------------------------------------
// Kernel 4: out_v = xv * weight[b][n], out_t = xt * weight[b][n]
// Pure HBM-bound (67 MB total): float4 loads/stores along n.
// ---------------------------------------------------------------------------
__global__ void nk_apply(const float* __restrict__ xv,
                         const float* __restrict__ xt,
                         const float* __restrict__ weight,
                         float* __restrict__ out)
{
    const size_t per4 = (size_t)BB * CC * NN / 4;           // float4s per tensor
    size_t gid = (size_t)blockIdx.x * blockDim.x + threadIdx.x;
    if (gid >= 2 * per4) return;
    const float* src = (gid < per4) ? xv : xt;
    const size_t id4 = (gid < per4) ? gid : gid - per4;
    const size_t e   = id4 * 4;                              // element index
    const int n = (int)(e % NN);
    const int b = (int)(e / ((size_t)CC * NN));

    float4 x = *(const float4*)(src + e);
    const float* w = weight + (size_t)b * NN + n;
    float4 o;
    o.x = x.x * w[0]; o.y = x.y * w[1]; o.z = x.z * w[2]; o.w = x.w * w[3];
    *(float4*)(out + gid * 4) = o;
}

// ---------------------------------------------------------------------------
extern "C" void kernel_launch(void* const* d_in, const int* in_sizes, int n_in,
                              void* d_out, int out_size, void* d_ws, size_t ws_size,
                              hipStream_t stream)
{
    (void)in_sizes; (void)n_in; (void)out_size; (void)ws_size;
    const float* xv = (const float*)d_in[0];
    const float* xt = (const float*)d_in[1];
    float* out = (float*)d_out;

    // Workspace layout: nv (8 MiB bf16) | nt (8 MiB bf16) | rel_mean | weight
    char* ws = (char*)d_ws;
    const size_t packBytes = (size_t)BB * NN * CC * sizeof(__bf16);
    __bf16* nv  = (__bf16*)ws;
    __bf16* nt  = (__bf16*)(ws + packBytes);
    float*  rel = (float*)(ws + 2 * packBytes);
    float*  wgt = rel + (size_t)BB * NN;

    nk_normalize_pack<<<dim3(NN / 256, BB, 2), 256, 0, stream>>>(xv, xt, nv, nt);
    nk_flash_rowstats<<<dim3(NN / 256, BB), 128, 0, stream>>>(nv, nt, rel);
    nk_weight<<<BB, 256, 0, stream>>>(rel, wgt);

    const size_t total4 = 2 * ((size_t)BB * CC * NN / 4);
    nk_apply<<<(unsigned)((total4 + 255) / 256), 256, 0, stream>>>(xv, xt, wgt, out);
}
